// CentroidModel_51762945851472
// MI455X (gfx1250) — matrix operand order repaired and checked by reading, not compiled, not run on hardware
//
#include <hip/hip_runtime.h>

// Problem constants (match reference)
#define B_SZ   512
#define T_SZ   256
#define D_SZ   64
#define H_SZ   256
#define G_SZ   1024   // 4*H
#define LAT_SZ 128
#define C_SZ   64

// Packed-K sizes for the two layers (K = Din + H), padded LDS strides
#define K0   (D_SZ + H_SZ)   // 320
#define K1   (H_SZ + H_SZ)   // 512
#define KP0  (K0 + 8)        // 328 bf16 -> 656B rows (16B aligned, bank-skewed)
#define KP1  (K1 + 8)        // 520 bf16 -> 1040B rows
#define KT0  (K0 / 32)       // 10 k-tiles (even)
#define KT1  (K1 / 32)       // 16 k-tiles (even)

typedef __bf16 bf16;
typedef __attribute__((ext_vector_type(16))) __bf16 v16bf;
typedef __attribute__((ext_vector_type(8)))  __bf16 v8bf;
typedef __attribute__((ext_vector_type(8)))  float   v8f;

union AFrag { v16bf v; v8bf h[2]; };

// Fast gate nonlinearities: one v_exp_f32 each, correct saturation.
__device__ __forceinline__ float sigm(float x)  { return 1.0f / (1.0f + __expf(-x)); }
__device__ __forceinline__ float tanh_f(float x){ return 1.0f - 2.0f / (__expf(2.0f * x) + 1.0f); }

// ---------------------------------------------------------------------------
// Fragment loaders.
// A-fragment lane layout (16-bit 16x32): lane<16 holds row=lane, K chunks
// {kt*32+0..7, kt*32+16..23}; lane>=16 holds K chunks {+8..15, +24..31}.
// B-fragment (16-bit 32x16): lane holds column N=lane&15, same K chunks as A.
// ---------------------------------------------------------------------------
__device__ __forceinline__ void loadA(const bf16* ldsA, int KP, int kt,
                                      int lane, AFrag& a)
{
  const int row  = lane & 15;
  const int ksel = (lane & 16) ? 8 : 0;
  a.h[0] = *(const v8bf*)&ldsA[row * KP + kt * 32 + ksel];
  a.h[1] = *(const v8bf*)&ldsA[row * KP + kt * 32 + ksel + 16];
}

__device__ __forceinline__ void loadB(const bf16* __restrict__ WT, int kt,
                                      int wave, int lane, v16bf (&b)[8])
{
  const int ncol = wave * 32 + (lane & 15);
  const int kh = (lane & 16) ? 8 : 0;
  #pragma unroll
  for (int g = 0; g < 4; ++g)
    #pragma unroll
    for (int j = 0; j < 2; ++j)
      #pragma unroll
      for (int s = 0; s < 16; ++s)
        b[g * 2 + j][s] = WT[(size_t)(kt * 32 + kh + (s < 8 ? s : s + 8)) * G_SZ + g * 256 + j * 16 + ncol];
}

__device__ __forceinline__ void mma8(const AFrag& a, const v16bf (&b)[8],
                                     v8f (&acc)[4][2])
{
  #pragma unroll
  for (int g = 0; g < 4; ++g)
    #pragma unroll
    for (int j = 0; j < 2; ++j)
      acc[g][j] = __builtin_amdgcn_wmma_f32_16x16x32_bf16(
          false, a.v, false, b[g * 2 + j], (short)0, acc[g][j], false, false);
  }

// ---------------------------------------------------------------------------
// Per-wave GEMM, 2-stage software pipeline with a bounded live set:
// exactly two B-fragment buffers (b0/b1) ever live; loop kept rolled
// (#pragma unroll 1) so the register allocator cannot widen the pipeline
// into spills. The b-refill for stage kt+2 issues before the b1 WMMA group
// and drains across the loop backedge.
// ---------------------------------------------------------------------------
template<int KTILES>
__device__ __forceinline__ void lstm_gemm(const bf16* ldsA, int KP,
                                          const bf16* __restrict__ WT,
                                          const float* __restrict__ biasC,
                                          v8f (&acc)[4][2], int wave, int lane)
{
  static_assert((KTILES & 1) == 0, "pipeline assumes even KTILES");
  const int nc = lane & 15;
  #pragma unroll
  for (int g = 0; g < 4; ++g) {
    #pragma unroll
    for (int j = 0; j < 2; ++j) {
      const float bv = biasC[g * 256 + wave * 32 + j * 16 + nc];
      #pragma unroll
      for (int r = 0; r < 8; ++r) acc[g][j][r] = bv;
    }
  }

  AFrag a;
  v16bf b0[8], b1[8];
  loadB(WT, 0, wave, lane, b0);

  #pragma unroll 1
  for (int kt = 0; kt < KTILES; kt += 2) {
    loadB(WT, kt + 1, wave, lane, b1);      // in flight under b0 WMMAs
    loadA(ldsA, KP, kt, lane, a);
    mma8(a, b0, acc);

    const int ktn = (kt + 2 < KTILES) ? kt + 2 : 0;  // clamp: branch-free refill
    loadB(WT, ktn, wave, lane, b0);         // in flight under b1 WMMAs + backedge
    loadA(ldsA, KP, kt + 1, lane, a);
    mma8(a, b1, acc);
  }
}

// ---------------------------------------------------------------------------
// Gate nonlinearities + cell update entirely in registers; scatter h (bf16)
// into next-step LDS operand(s) and optionally global memory.
// C/D f32 16x16 layout: element r of VGPR <-> row r + 8*(lane>=16), col lane&15.
// ---------------------------------------------------------------------------
__device__ __forceinline__ void lstm_epi(v8f (&acc)[4][2], v8f (&cst)[2],
                                         bf16* lds1, int s1, int c1off,
                                         bf16* lds2, int s2, int c2off,
                                         bf16* gptr, bool doG,
                                         int wave, int lane)
{
  const int rbase = (lane & 16) ? 8 : 0;
  const int nc = lane & 15;
  #pragma unroll
  for (int j = 0; j < 2; ++j) {
    const int hcol = wave * 32 + j * 16 + nc;
    #pragma unroll
    for (int r = 0; r < 8; ++r) {
      const float iv = sigm(acc[0][j][r]);
      const float fv = sigm(acc[1][j][r]);
      const float gv = tanh_f(acc[2][j][r]);
      const float ov = sigm(acc[3][j][r]);
      const float cc = fv * cst[j][r] + iv * gv;
      cst[j][r] = cc;
      const float hv = ov * tanh_f(cc);
      const bf16 hb = (bf16)hv;
      const int row = r + rbase;
      lds1[row * s1 + c1off + hcol] = hb;
      if (lds2) lds2[row * s2 + c2off + hcol] = hb;
      if (doG)  gptr[row * H_SZ + hcol] = hb;
    }
  }
}

// ---------------------------------------------------------------------------
// Fused 2-layer LSTM, persistent over all T. One block = 16 batch rows.
// LDS A0 = [16][KP0] (x_t | h_l0), A1 = [16][KP1] (h_l0 | h_l1). ~27KB LDS.
// Only 32 blocks exist; occupancy is irrelevant -> launch_bounds(256,1)
// gives the allocator the full register budget (no spills).
// ---------------------------------------------------------------------------
__global__ __launch_bounds__(256, 1) void lstm2_fused_kernel(
    const float* __restrict__ x,
    const bf16* __restrict__ WT0, const bf16* __restrict__ WT1,
    const float* __restrict__ bias0, const float* __restrict__ bias1,
    bf16* __restrict__ h2last)
{
  __shared__ __align__(32) bf16 A0[16 * KP0];
  __shared__ __align__(32) bf16 A1[16 * KP1];

  const int tid    = threadIdx.x;
  const int wave   = tid >> 5;   // wave32
  const int lane   = tid & 31;
  const int wgBase = blockIdx.x * 16;

  for (int i = tid; i < 16 * KP0; i += 256) A0[i] = (bf16)0.0f;
  for (int i = tid; i < 16 * KP1; i += 256) A1[i] = (bf16)0.0f;

  v8f c0[2], c1[2];
  #pragma unroll
  for (int j = 0; j < 2; ++j) {
    #pragma unroll
    for (int r = 0; r < 8; ++r) { c0[j][r] = 0.0f; c1[j][r] = 0.0f; }
  }
  __syncthreads();

  v8f acc[4][2];
  #pragma unroll 1
  for (int t = 0; t < T_SZ; ++t) {
    // stage x_t (fp32 -> bf16) into A0 cols [0, 64): 1024 elems / 256 thr
    for (int i = tid; i < 16 * D_SZ; i += 256) {
      const int row = i >> 6, col = i & 63;
      A0[row * KP0 + col] =
          (bf16)x[((size_t)(wgBase + row) * T_SZ + t) * D_SZ + col];
    }
    __syncthreads();                                   // x + prev h_l1 visible

    lstm_gemm<KT0>(A0, KP0, WT0, bias0, acc, wave, lane);
    __syncthreads();                                   // A0 reads done
    lstm_epi(acc, c0, A0, KP0, D_SZ,                   // h_l0 -> A0 h-region
             A1, KP1, 0,                               //       -> A1 x-region
             (bf16*)nullptr, false, wave, lane);
    __syncthreads();                                   // h_l0 visible in A1

    lstm_gemm<KT1>(A1, KP1, WT1, bias1, acc, wave, lane);
    __syncthreads();                                   // A1 reads done
    lstm_epi(acc, c1, A1, KP1, H_SZ,                   // h_l1 -> A1 h-region
             (bf16*)nullptr, 0, 0,
             h2last + (size_t)wgBase * H_SZ, (t == T_SZ - 1), wave, lane);
  }
}

// ---------------------------------------------------------------------------
// Pack weights: WT_l[k][n] = W^T concat along K, fp32 -> bf16; fuse biases.
// ---------------------------------------------------------------------------
__global__ void prep_kernel(
    const float* __restrict__ Wih0, const float* __restrict__ Whh0,
    const float* __restrict__ bih0, const float* __restrict__ bhh0,
    const float* __restrict__ Wih1, const float* __restrict__ Whh1,
    const float* __restrict__ bih1, const float* __restrict__ bhh1,
    bf16* __restrict__ WT0, bf16* __restrict__ WT1,
    float* __restrict__ bias0, float* __restrict__ bias1)
{
  const int N0 = K0 * G_SZ;                    // 327680
  const int N1 = K1 * G_SZ;                    // 524288
  const int NT = N0 + N1 + 2 * G_SZ;
  for (int i = blockIdx.x * blockDim.x + threadIdx.x; i < NT;
       i += gridDim.x * blockDim.x) {
    if (i < N0) {
      const int k = i / G_SZ, n = i % G_SZ;
      const float v = (k < D_SZ) ? Wih0[n * D_SZ + k]
                                 : Whh0[n * H_SZ + (k - D_SZ)];
      WT0[i] = (bf16)v;
    } else if (i < N0 + N1) {
      const int j = i - N0;
      const int k = j / G_SZ, n = j % G_SZ;
      const float v = (k < H_SZ) ? Wih1[n * H_SZ + k]
                                 : Whh1[n * H_SZ + (k - H_SZ)];
      WT1[j] = (bf16)v;
    } else {
      const int j = i - N0 - N1;
      if (j < G_SZ) bias0[j] = bih0[j] + bhh0[j];
      else          bias1[j - G_SZ] = bih1[j - G_SZ] + bhh1[j - G_SZ];
    }
  }
}

// ---------------------------------------------------------------------------
// Head: emb = h2_last @ We^T + be; dists to centroids; argmin + softmax.
// One block per batch row; compute-trivial (~17 MFLOP total), plain fp32.
// Output layout: out[0..B) = decoded (as float), out[B + b*C + i] = probs.
// ---------------------------------------------------------------------------
__global__ __launch_bounds__(128) void head_kernel(
    const bf16* __restrict__ h2last, const float* __restrict__ We,
    const float* __restrict__ be, const float* __restrict__ centroids,
    const int* __restrict__ class_nums, float* __restrict__ out)
{
  __shared__ float emb[LAT_SZ];
  __shared__ float dist[C_SZ];
  const int b = blockIdx.x;
  const int t = threadIdx.x;

  const bf16*  hrow = h2last + (size_t)b * H_SZ;
  const float* wrow = We + (size_t)t * H_SZ;
  float s = be[t];
  for (int k = 0; k < H_SZ; ++k) s += (float)hrow[k] * wrow[k];
  emb[t] = s;
  __syncthreads();

  if (t < C_SZ) {
    const float* crow = centroids + (size_t)t * LAT_SZ;
    float d = 0.0f;
    for (int l = 0; l < LAT_SZ; ++l) { const float df = emb[l] - crow[l]; d += df * df; }
    dist[t] = d;
  }
  __syncthreads();

  if (t == 0) {
    float mn = dist[0]; int arg = 0;
    for (int i = 1; i < C_SZ; ++i) if (dist[i] < mn) { mn = dist[i]; arg = i; }
    out[b] = (float)class_nums[arg];
    float sum = 0.0f;
    for (int i = 0; i < C_SZ; ++i) sum += __expf(mn - dist[i]);
    const float inv = 1.0f / sum;
    for (int i = 0; i < C_SZ; ++i)
      out[B_SZ + (size_t)b * C_SZ + i] = __expf(mn - dist[i]) * inv;
  }
}

// ---------------------------------------------------------------------------
extern "C" void kernel_launch(void* const* d_in, const int* in_sizes, int n_in,
                              void* d_out, int out_size, void* d_ws, size_t ws_size,
                              hipStream_t stream) {
  (void)in_sizes; (void)n_in; (void)out_size; (void)ws_size;
  const float* x     = (const float*)d_in[0];
  const float* Wih0  = (const float*)d_in[1];
  const float* Whh0  = (const float*)d_in[2];
  const float* bih0  = (const float*)d_in[3];
  const float* bhh0  = (const float*)d_in[4];
  const float* Wih1  = (const float*)d_in[5];
  const float* Whh1  = (const float*)d_in[6];
  const float* bih1  = (const float*)d_in[7];
  const float* bhh1  = (const float*)d_in[8];
  const float* We    = (const float*)d_in[9];
  const float* be    = (const float*)d_in[10];
  const float* cents = (const float*)d_in[11];
  const int*   cnums = (const int*)d_in[12];

  // Workspace carve (~2.0 MB total, all offsets 32B-aligned)
  char*  ws    = (char*)d_ws;
  float* bias0 = (float*)(ws + 0);
  float* bias1 = (float*)(ws + 4096);
  bf16*  WT0   = (bf16*)(ws + 8192);
  bf16*  WT1   = (bf16*)(ws + 8192 + (size_t)K0 * G_SZ * 2);
  bf16*  h2l   = (bf16*)(ws + 8192 + (size_t)K0 * G_SZ * 2 + (size_t)K1 * G_SZ * 2);

  prep_kernel<<<512, 256, 0, stream>>>(Wih0, Whh0, bih0, bhh0,
                                       Wih1, Whh1, bih1, bhh1,
                                       WT0, WT1, bias0, bias1);
  lstm2_fused_kernel<<<B_SZ / 16, 256, 0, stream>>>(x, WT0, WT1, bias0, bias1, h2l);
  head_kernel<<<B_SZ, 128, 0, stream>>>(h2l, We, be, cents, cnums, (float*)d_out);
}
